// ConvLayer_74028056314444
// MI455X (gfx1250) — compile-verified, matrix operand order checked by reading, not run on hardware
//
#include <hip/hip_runtime.h>
#include <hip/hip_bf16.h>

// ---------------------------------------------------------------------------
// GAT-style layer for MI455X (gfx1250, wave32, WMMA).
//   N=50000 nodes, 16 neighbors each, 128->128 linear, attention, softmax, agg.
// Phase 1: self_lin = self_emb@W + b  (WMMA bf16), self-score = self_lin@w1.
// Phase 2: per node, neigh tile (16x128) GEMM in registers, scores, softmax,
//          weighted aggregation, ReLU output. neigh_lin never hits memory.
// ---------------------------------------------------------------------------

typedef __attribute__((ext_vector_type(16))) __bf16 v16bf;
typedef __attribute__((ext_vector_type(8)))  __bf16 v8bf;
typedef __attribute__((ext_vector_type(8)))  float  v8f;

#define DIM       128
#define NEIGH     16
#define WT_STRIDE 136   // padded LDS row stride (bf16 elems): 272B -> lanes hit distinct banks

// f32 -> bf16, round-to-nearest-even (bit trick; no dependence on cvt builtins)
__device__ __forceinline__ __bf16 f2bf(float f) {
    unsigned u = __builtin_bit_cast(unsigned, f);
    u = (u + 0x7FFFu + ((u >> 16) & 1u)) >> 16;
    unsigned short s = (unsigned short)u;
    return __builtin_bit_cast(__bf16, s);
}

// Copy W (k-major [128][128] f32) into LDS as bf16, transposed (n-major, padded).
__device__ __forceinline__ void fill_wt(__bf16* sWt, const float* __restrict__ W) {
    for (int idx = threadIdx.x; idx < DIM * DIM; idx += blockDim.x) {
        int k = idx >> 7, n = idx & 127;
        sWt[n * WT_STRIDE + k] = f2bf(W[idx]);
    }
    __syncthreads();
}

// A fragment (16x32 bf16): lane m=(l&15) holds K in {kb..kb+7, kb+16..kb+23},
// kb = 8*(l>>4). Source row is f32; two 32B-aligned float4 pairs per lane.
__device__ __forceinline__ v16bf load_a_frag(const float* __restrict__ row, int lane, int ks) {
    int kb = ks * 32 + ((lane >> 4) << 3);
    const float4* p = reinterpret_cast<const float4*>(row + kb);
    const float4* q = reinterpret_cast<const float4*>(row + kb + 16);
    float4 x0 = p[0], x1 = p[1], y0 = q[0], y1 = q[1];
    v16bf a;
    a[0]  = f2bf(x0.x); a[1]  = f2bf(x0.y); a[2]  = f2bf(x0.z); a[3]  = f2bf(x0.w);
    a[4]  = f2bf(x1.x); a[5]  = f2bf(x1.y); a[6]  = f2bf(x1.z); a[7]  = f2bf(x1.w);
    a[8]  = f2bf(y0.x); a[9]  = f2bf(y0.y); a[10] = f2bf(y0.z); a[11] = f2bf(y0.w);
    a[12] = f2bf(y1.x); a[13] = f2bf(y1.y); a[14] = f2bf(y1.z); a[15] = f2bf(y1.w);
    return a;
}

// B fragment (32x16 bf16), mirrored layout: lane n=(l&15) holds column n,
// same K split as A. W^T in LDS makes each half an aligned 16B ds_load_b128.
__device__ __forceinline__ v16bf load_b_frag(const __bf16* sWt, int lane, int nt, int ks) {
    int n  = nt * 16 + (lane & 15);
    int kb = ks * 32 + ((lane >> 4) << 3);
    const __bf16* base = sWt + n * WT_STRIDE + kb;
    v8bf lo = *reinterpret_cast<const v8bf*>(base);        // K = kb .. kb+7
    v8bf hi = *reinterpret_cast<const v8bf*>(base + 16);   // K = kb+16 .. kb+23
    v16bf b;
#pragma unroll
    for (int i = 0; i < 8; ++i) { b[i] = lo[i]; b[8 + i] = hi[i]; }
    return b;
}

// ---------------------------------------------------------------------------
// Phase 1: self_lin and self attention score (per 16-node tile, one wave each)
// ---------------------------------------------------------------------------
__global__ __launch_bounds__(256) void gat_self_kernel(
    const float* __restrict__ self_emb, const float* __restrict__ W,
    const float* __restrict__ b, const float* __restrict__ w_att,
    float* __restrict__ ws_selfLin, float* __restrict__ ws_sscore,
    int n_nodes, int n_tiles)
{
    __shared__ __bf16 sWt[DIM * WT_STRIDE];
    fill_wt(sWt, W);

    const int wave = threadIdx.x >> 5;
    const int lane = threadIdx.x & 31;
    const int tile = blockIdx.x * 8 + wave;
    if (tile >= n_tiles) return;

    const int half = lane >> 4;
    const int lnib = lane & 15;

    int mrow = tile * 16 + lnib;
    if (mrow >= n_nodes) mrow = n_nodes - 1;   // clamp; stores are guarded below
    const float* Arow = self_emb + (size_t)mrow * DIM;

    v8f acc[8];
#pragma unroll
    for (int nt = 0; nt < 8; ++nt) {
        float bv = b[nt * 16 + lnib];
#pragma unroll
        for (int v = 0; v < 8; ++v) acc[nt][v] = bv;
    }

#pragma unroll
    for (int ks = 0; ks < 4; ++ks) {
        v16bf a = load_a_frag(Arow, lane, ks);
#pragma unroll
        for (int nt = 0; nt < 8; ++nt) {
            v16bf bb = load_b_frag(sWt, lane, nt, ks);
            acc[nt] = __builtin_amdgcn_wmma_f32_16x16x32_bf16(
                false, a, false, bb, (short)0, acc[nt], false, false);
        }
    }

    // self score: sscore[m] = sum_n self_lin[m][n] * w1[n]
    float w1c[8];
#pragma unroll
    for (int nt = 0; nt < 8; ++nt) w1c[nt] = w_att[nt * 16 + lnib];
    float part[8];
#pragma unroll
    for (int v = 0; v < 8; ++v) {
        float s = 0.f;
#pragma unroll
        for (int nt = 0; nt < 8; ++nt) s += acc[nt][v] * w1c[nt];
#pragma unroll
        for (int mask = 1; mask < 16; mask <<= 1) s += __shfl_xor(s, mask, 32);
        part[v] = s;   // every lane in half h now has score of row m = 8h+v
    }

    // store self_lin (row-major f32) for phase 2
#pragma unroll
    for (int v = 0; v < 8; ++v) {
        int m = tile * 16 + half * 8 + v;
        if (m < n_nodes) {
            float* dst = ws_selfLin + (size_t)m * DIM + lnib;
#pragma unroll
            for (int nt = 0; nt < 8; ++nt) dst[nt * 16] = acc[nt][v];
        }
    }

    // one lane per row writes the score
    float sv = part[0];
#pragma unroll
    for (int v = 1; v < 8; ++v) if (lnib == v) sv = part[v];
    int m0 = tile * 16 + half * 8 + (lnib & 7);
    if (lnib < 8 && m0 < n_nodes) ws_sscore[m0] = sv;
}

// ---------------------------------------------------------------------------
// Phase 2: per node (one wave): neigh GEMM tile -> scores -> softmax -> agg
// ---------------------------------------------------------------------------
__global__ __launch_bounds__(256) void gat_neigh_kernel(
    const float* __restrict__ neigh_emb, const float* __restrict__ W,
    const float* __restrict__ b, const float* __restrict__ w_att,
    const float* __restrict__ b_att,
    const float* __restrict__ ws_selfLin, const float* __restrict__ ws_sscore,
    float* __restrict__ out_emb, float* __restrict__ out_att, int n_nodes)
{
    __shared__ __bf16 sWt[DIM * WT_STRIDE];
    fill_wt(sWt, W);

    const int wave = threadIdx.x >> 5;
    const int lane = threadIdx.x & 31;
    const int node = blockIdx.x * 8 + wave;
    if (node >= n_nodes) return;

    const int half = lane >> 4;
    const int lnib = lane & 15;

    // 16 neighbors of this node = one contiguous 16x128 A tile
    const float* Arow = neigh_emb + ((size_t)node * NEIGH + lnib) * DIM;

    v8f acc[8];
#pragma unroll
    for (int nt = 0; nt < 8; ++nt) {
        float bv = b[nt * 16 + lnib];
#pragma unroll
        for (int v = 0; v < 8; ++v) acc[nt][v] = bv;
    }

#pragma unroll
    for (int ks = 0; ks < 4; ++ks) {
        v16bf a = load_a_frag(Arow, lane, ks);
#pragma unroll
        for (int nt = 0; nt < 8; ++nt) {
            v16bf bb = load_b_frag(sWt, lane, nt, ks);
            acc[nt] = __builtin_amdgcn_wmma_f32_16x16x32_bf16(
                false, a, false, bb, (short)0, acc[nt], false, false);
        }
    }

    // attention scores: score[m] = neigh_lin[m]@w2 + self_score + b_att
    float w2c[8];
#pragma unroll
    for (int nt = 0; nt < 8; ++nt) w2c[nt] = w_att[DIM + nt * 16 + lnib];
    const float sb = ws_sscore[node] + b_att[0];

    float e[8];
    float esum = 0.f;
#pragma unroll
    for (int v = 0; v < 8; ++v) {
        float s = 0.f;
#pragma unroll
        for (int nt = 0; nt < 8; ++nt) s += acc[nt][v] * w2c[nt];
#pragma unroll
        for (int mask = 1; mask < 16; mask <<= 1) s += __shfl_xor(s, mask, 32);
        s += sb;
        float g = 1.0f / (1.0f + __expf(-s));   // sigmoid; output in (0,1)
        e[v] = __expf(g);                        // softmax numerator (stable: g bounded)
        esum += e[v];
    }
    float total = esum + __shfl_xor(esum, 16, 32);   // merge the two M-halves
    float inv   = 1.0f / total;
    float attv[8];
#pragma unroll
    for (int v = 0; v < 8; ++v) attv[v] = e[v] * inv;

    // weighted aggregation + fused output: relu(0.5*self_lin + 0.5*agg)
#pragma unroll
    for (int nt = 0; nt < 8; ++nt) {
        float s = 0.f;
#pragma unroll
        for (int v = 0; v < 8; ++v) s += attv[v] * acc[nt][v];
        s += __shfl_xor(s, 16, 32);              // sum over both M-halves
        if (lane < 16) {
            float sl = ws_selfLin[(size_t)node * DIM + nt * 16 + lnib];
            float o  = 0.5f * sl + 0.5f * s;
            out_emb[(size_t)node * DIM + nt * 16 + lnib] = fmaxf(o, 0.0f);
        }
    }

    // write attention weights (one lane per neighbor)
    float av = attv[0];
#pragma unroll
    for (int v = 1; v < 8; ++v) if (lnib == v) av = attv[v];
    if (lnib < 8) out_att[(size_t)node * NEIGH + half * 8 + lnib] = av;
}

// ---------------------------------------------------------------------------
extern "C" void kernel_launch(void* const* d_in, const int* in_sizes, int n_in,
                              void* d_out, int out_size, void* d_ws, size_t ws_size,
                              hipStream_t stream) {
    const float* self_emb  = (const float*)d_in[0];
    const float* neigh_emb = (const float*)d_in[1];
    const float* W         = (const float*)d_in[2];
    const float* b         = (const float*)d_in[3];
    const float* w_att     = (const float*)d_in[4];
    const float* b_att     = (const float*)d_in[5];

    const int n_nodes = in_sizes[0] / DIM;

    float* ws_selfLin = (float*)d_ws;                       // n_nodes * 128 f32
    float* ws_sscore  = ws_selfLin + (size_t)n_nodes * DIM; // n_nodes f32

    float* out_emb = (float*)d_out;                         // n_nodes * 128
    float* out_att = out_emb + (size_t)n_nodes * DIM;       // n_nodes * 16

    const int n_tiles = (n_nodes + 15) / 16;
    dim3 blk(256);

    gat_self_kernel<<<(n_tiles + 7) / 8, blk, 0, stream>>>(
        self_emb, W, b, w_att, ws_selfLin, ws_sscore, n_nodes, n_tiles);

    gat_neigh_kernel<<<(n_nodes + 7) / 8, blk, 0, stream>>>(
        neigh_emb, W, b, w_att, b_att, ws_selfLin, ws_sscore,
        out_emb, out_att, n_nodes);
}
